// NGDCSDE_32933809225770
// MI455X (gfx1250) — compile-verified, matrix operand order checked by reading, not run on hardware
//
#include <hip/hip_runtime.h>
#include <math.h>

// ---------------- dims ----------------
#define BB   4
#define TT   33
#define NN   500          // real node count
#define NP   512          // padded node count (multiple of 32)
#define DIN  64
#define DRNN 128
#define DST  64
#define MP   (BB * NP)    // 2048 padded rows for [B*N, d] GEMMs
#define DT_  0.5f
#define SQDT 0.70710678118654752f

typedef __attribute__((ext_vector_type(16))) _Float16 v16h;
typedef __attribute__((ext_vector_type(8)))  _Float16 v8h;
typedef __attribute__((ext_vector_type(8)))  float    v8f;

__device__ __forceinline__ float actf(float x, int act) {
  if (act == 1) return tanhf(x);
  if (act == 2) return 1.0f / (1.0f + expf(-x));
  return x;
}

#define SHUF16(lo, hi) __builtin_shufflevector(lo, hi, 0,1,2,3,4,5,6,7,8,9,10,11,12,13,14,15)

// ---------------------------------------------------------------------------
// WMMA GEMM, one wave computes a 16(M) x 64(N) tile.
//   A16: row-major f16 [M,K] (lda, per-z stride sA)        -> A fragments
//   B16: col-major f16 [N,K] (ldb, per-z stride sB)        -> B fragments
// K loop: unroll-by-2 ping-pong register pipeline (X=even tiles, Y=odd tiles;
// consume X, refill X with tile k+2, consume Y, refill Y with tile k+3).
// Stable register names -> no rotation movs; s_wait_loadcnt before a WMMA
// group only waits on loads issued a full group earlier. nT is always even
// for every K used here (64..576), so there is no remainder path.
// Outputs (each optional):
//   D    f32  row-major (bias/act/residual epilogue), store guarded by Mstore
//   R16  f16  row-major  (feeds next GEMM's A operand)
//   C16  f16  col-major  (feeds next GEMM's B operand); row index decomposed
//        with period cPer:  addr = z*sCz + (m/cPer)*sCm + col*ldc + (m%cPer)
// All dims padded: K%32==0, M%16==0, N%64==0 -> zero branches around loads.
// ---------------------------------------------------------------------------
__global__ __launch_bounds__(32)
void k_gemm(const _Float16* __restrict__ A16, int lda, long long sA,
            const _Float16* __restrict__ B16, int ldb, long long sB,
            const float* __restrict__ bias,
            const float* __restrict__ R, int ldr, long long sR, float resScale,
            float* __restrict__ D, int ldd, long long sD, int Mstore,
            _Float16* __restrict__ R16, int ld16, long long s16,
            _Float16* __restrict__ C16, int ldc, int cPer, long long sCz, long long sCm,
            int Kk, int act, float outScale)
{
  const int lane = threadIdx.x & 31;
  const int half = lane >> 4;          // K-half selector inside fragments
  const int l16  = lane & 15;
  const int m0 = blockIdx.y << 4;
  const int n0 = blockIdx.x << 6;      // 64 columns per wave
  const int b  = blockIdx.z;

  const _Float16* Ar = A16 + (long long)b * sA + (long long)(m0 + l16) * lda;
  const _Float16* B0 = B16 + (long long)b * sB + (long long)(n0 + l16) * ldb;
  const _Float16* Bp[4];
#pragma unroll
  for (int j = 0; j < 4; ++j) Bp[j] = B0 + (long long)(j * 16) * ldb;

  const int o0 = half * 8;             // fragment sub-offsets within a K tile
  const int o1 = 16 + half * 8;
  const int nT = Kk >> 5;              // number of K tiles (even for all K used)

  // prologue: X <- tile 0, Y <- tile 1
  v8h xa0 = *(const v8h*)(Ar + o0);
  v8h xa1 = *(const v8h*)(Ar + o1);
  v8h xb0[4], xb1[4];
#pragma unroll
  for (int j = 0; j < 4; ++j) {
    xb0[j] = *(const v8h*)(Bp[j] + o0);
    xb1[j] = *(const v8h*)(Bp[j] + o1);
  }
  v8h ya0 = *(const v8h*)(Ar + 32 + o0);
  v8h ya1 = *(const v8h*)(Ar + 32 + o1);
  v8h yb0[4], yb1[4];
#pragma unroll
  for (int j = 0; j < 4; ++j) {
    yb0[j] = *(const v8h*)(Bp[j] + 32 + o0);
    yb1[j] = *(const v8h*)(Bp[j] + 32 + o1);
  }

  v8f acc[4] = {{}, {}, {}, {}};
  for (int it = 0; it < nT; it += 2) {
    // ---- consume X (tile it) ----
    {
      v16h af = SHUF16(xa0, xa1);
#pragma unroll
      for (int j = 0; j < 4; ++j) {
        v16h bf = SHUF16(xb0[j], xb1[j]);
        acc[j] = __builtin_amdgcn_wmma_f32_16x16x32_f16(false, af, false, bf,
                                                        (short)0, acc[j], false, false);
      }
    }
    // ---- refill X with tile it+2 ----
    if (it + 2 < nT) {
      const int k = (it + 2) << 5;
      __builtin_prefetch(Ar + k + 64, 0, 1);
      xa0 = *(const v8h*)(Ar + k + o0);
      xa1 = *(const v8h*)(Ar + k + o1);
#pragma unroll
      for (int j = 0; j < 4; ++j) {
        xb0[j] = *(const v8h*)(Bp[j] + k + o0);
        xb1[j] = *(const v8h*)(Bp[j] + k + o1);
      }
    }
    // ---- consume Y (tile it+1) ----
    {
      v16h af = SHUF16(ya0, ya1);
#pragma unroll
      for (int j = 0; j < 4; ++j) {
        v16h bf = SHUF16(yb0[j], yb1[j]);
        acc[j] = __builtin_amdgcn_wmma_f32_16x16x32_f16(false, af, false, bf,
                                                        (short)0, acc[j], false, false);
      }
    }
    // ---- refill Y with tile it+3 ----
    if (it + 3 < nT) {
      const int k = (it + 3) << 5;
      ya0 = *(const v8h*)(Ar + k + o0);
      ya1 = *(const v8h*)(Ar + k + o1);
#pragma unroll
      for (int j = 0; j < 4; ++j) {
        yb0[j] = *(const v8h*)(Bp[j] + k + o0);
        yb1[j] = *(const v8h*)(Bp[j] + k + o1);
      }
    }
  }

  // ---------------- epilogue ----------------
  const float* Rp = R ? (R + (long long)b * sR) : nullptr;
  float*       Dp = D ? (D + (long long)b * sD) : nullptr;
  _Float16*  R16p = R16 ? (R16 + (long long)b * s16) : nullptr;
  const int mb = m0 + 8 * half;        // C/D layout: element v -> row mb+v
#pragma unroll
  for (int j = 0; j < 4; ++j) {
    const int col = n0 + j * 16 + l16;
    const float bval = bias ? bias[col] : 0.0f;
    float vals[8];
#pragma unroll
    for (int v = 0; v < 8; ++v) {
      float x = actf(acc[j][v] + bval, act) * outScale;
      if (Rp) x += resScale * Rp[(long long)(mb + v) * ldr + col];
      vals[v] = x;
    }
    if (Dp) {
#pragma unroll
      for (int v = 0; v < 8; ++v)
        if (mb + v < Mstore) Dp[(long long)(mb + v) * ldd + col] = vals[v];
    }
    if (R16p) {
#pragma unroll
      for (int v = 0; v < 8; ++v)
        R16p[(long long)(mb + v) * ld16 + col] = (_Float16)vals[v];
    }
    if (C16) {
      v8h pk;
#pragma unroll
      for (int v = 0; v < 8; ++v) pk[v] = (_Float16)vals[v];
      int bb = mb / cPer, nn2 = mb % cPer;
      _Float16* cp = C16 + (long long)b * sCz + (long long)bb * sCm
                         + (long long)col * ldc + nn2;
      *(v8h*)cp = pk;                  // 16B aligned: nn2 % 8 == 0
    }
  }
}

// ---------------- conversion / pack / elementwise kernels ----------------
// zero-padded f16 adjacency [NP,NP]
__global__ void k_adj16(_Float16* __restrict__ dst, const float* __restrict__ adj) {
  int i = blockIdx.x * blockDim.x + threadIdx.x;
  if (i >= NP * NP) return;
  int r = i >> 9, c = i & 511;
  dst[i] = (r < NN && c < NN) ? (_Float16)adj[r * NN + c] : (_Float16)0.0f;
}
// W [K,N] f32 -> col-major f16 [N,K]
__global__ void k_w16t(_Float16* __restrict__ dst, const float* __restrict__ src,
                       int K, int N) {
  int i = blockIdx.x * blockDim.x + threadIdx.x;
  if (i >= K * N) return;
  int n = i / K, k = i % K;
  dst[i] = (_Float16)src[(long long)k * N + n];
}
// straight f32 -> f16 copy (GRU weights: Wih itself is already B col-major)
__global__ void k_w16c(_Float16* __restrict__ dst, const float* __restrict__ src, int n) {
  int i = blockIdx.x * blockDim.x + threadIdx.x;
  if (i < n) dst[i] = (_Float16)src[i];
}
__global__ void k_fill32(float* __restrict__ dst, float v, int n) {
  int i = blockIdx.x * blockDim.x + threadIdx.x;
  if (i < n) dst[i] = v;
}
__global__ void k_fill16(_Float16* __restrict__ dst, int n) {
  int i = blockIdx.x * blockDim.x + threadIdx.x;
  if (i < n) dst[i] = (_Float16)0.0f;
}
__global__ void k_broadcast(float* __restrict__ dst, const float* __restrict__ vec,
                            int M, int Dd) {
  int i = blockIdx.x * blockDim.x + threadIdx.x;
  if (i < M * Dd) dst[i] = vec[i % Dd];
}
// f32 [MP,128] -> XC slot0 (row f16, ld) + XT col-major f16 [B][128][NP]; pads = 0
__global__ void k_pack_h(_Float16* __restrict__ xc, int ld, _Float16* __restrict__ xt,
                         const float* __restrict__ src) {
  int i = blockIdx.x * blockDim.x + threadIdx.x;
  if (i >= MP * DRNN) return;
  int m = i / DRNN, d = i % DRNN;
  int b = m >> 9, n = m & 511;
  float v = (n < NN) ? src[(long long)m * DRNN + d] : 0.0f;
  _Float16 h = (_Float16)v;
  xc[(long long)m * ld + d] = h;
  xt[((long long)b * DRNN + d) * NP + n] = h;
}
// slot0 = [x(64) | h(128)] at time t  (x = values*masks); pads = 0
__global__ void k_pack_xh(_Float16* __restrict__ xc, int ld, _Float16* __restrict__ xt,
                          const float* __restrict__ values,
                          const float* __restrict__ masks,
                          const float* __restrict__ h, int t) {
  int i = blockIdx.x * blockDim.x + threadIdx.x;
  if (i >= MP * 192) return;
  int m = i / 192, d = i % 192;
  int b = m >> 9, n = m & 511;
  float v = 0.0f;
  if (n < NN) {
    if (d < DIN) {
      long long idx = (((long long)b * TT + t) * NN + n) * DIN + d;
      v = values[idx] * masks[idx];
    } else {
      v = h[(long long)m * DRNN + (d - DIN)];
    }
  }
  _Float16 hh = (_Float16)v;
  xc[(long long)m * ld + d] = hh;
  xt[((long long)b * 192 + d) * NP + n] = hh;
}
// slot0 = [x(64) | r*h(128)], r = ru[:,0:128]
__global__ void k_pack_xrh(_Float16* __restrict__ xc, int ld, _Float16* __restrict__ xt,
                           const float* __restrict__ values,
                           const float* __restrict__ masks,
                           const float* __restrict__ ru,
                           const float* __restrict__ h, int t) {
  int i = blockIdx.x * blockDim.x + threadIdx.x;
  if (i >= MP * 192) return;
  int m = i / 192, d = i % 192;
  int b = m >> 9, n = m & 511;
  float v = 0.0f;
  if (n < NN) {
    if (d < DIN) {
      long long idx = (((long long)b * TT + t) * NN + n) * DIN + d;
      v = values[idx] * masks[idx];
    } else {
      int dd = d - DIN;
      v = ru[(long long)m * 256 + dd] * h[(long long)m * DRNN + dd];
    }
  }
  _Float16 hh = (_Float16)v;
  xc[(long long)m * ld + d] = hh;
  xt[((long long)b * 192 + d) * NP + n] = hh;
}
__global__ void k_om(float* __restrict__ om, const float* __restrict__ masks, int t) {
  int m = blockIdx.x * blockDim.x + threadIdx.x;
  if (m >= MP) return;
  int b = m >> 9, n = m & 511;
  float o = 0.0f;
  if (n < NN) {
    long long base = (((long long)b * TT + t) * NN + n) * DIN;
    float s = 0.0f;
    for (int d = 0; d < DIN; ++d) s += fabsf(masks[base + d]);
    o = (s > 1e-4f) ? 1.0f : 0.0f;
  }
  om[m] = o;
}
// out = h1*(1-om) + om*( u*h1 + (1-u)*c ),  u = ru[:,128:256]
__global__ void k_gru_blend(float* __restrict__ out, const float* __restrict__ h1,
                            const float* __restrict__ ru, const float* __restrict__ cc,
                            const float* __restrict__ om) {
  int i = blockIdx.x * blockDim.x + threadIdx.x;
  if (i >= MP * DRNN) return;
  int m = i / DRNN, d = i % DRNN;
  float u = ru[(long long)m * 256 + 128 + d];
  float h = h1[i], c = cc[i], o = om[m];
  out[i] = h * (1.0f - o) + o * (u * h + (1.0f - u) * c);
}
// dst16 = (f16)(a*b)   (gated h_pre)
__global__ void k_mul16(_Float16* __restrict__ dst, const float* __restrict__ a,
                        const float* __restrict__ b, int n) {
  int i = blockIdx.x * blockDim.x + threadIdx.x;
  if (i < n) dst[i] = (_Float16)(a[i] * b[i]);
}
// torch GRU pointwise; updates h (f32 + f16) and stores to ze16
__global__ void k_gru_cell(float* __restrict__ h, _Float16* __restrict__ h16,
                           _Float16* __restrict__ ze16,
                           const float* __restrict__ gi, const float* __restrict__ gh) {
  int i = blockIdx.x * blockDim.x + threadIdx.x;
  if (i >= MP * DRNN) return;
  int m = i / DRNN, d = i % DRNN;
  long long bq = (long long)m * 384;
  float r = 1.0f / (1.0f + expf(-(gi[bq + d]       + gh[bq + d])));
  float z = 1.0f / (1.0f + expf(-(gi[bq + 128 + d] + gh[bq + 128 + d])));
  float g = tanhf(gi[bq + 256 + d] + r * gh[bq + 256 + d]);
  float h2 = (1.0f - z) * g + z * h[i];
  h[i] = h2;
  h16[i] = (_Float16)h2;
  ze16[i] = (_Float16)h2;
}
// sin16 = (f16)( prior * (zdiff*Z)^2 )
__global__ void k_sde_in(_Float16* __restrict__ u, const float* __restrict__ pri,
                         const float* __restrict__ zdf, const float* __restrict__ z,
                         int n) {
  int i = blockIdx.x * blockDim.x + threadIdx.x;
  if (i >= n) return;
  float q = zdf[i] * z[i];
  u[i] = (_Float16)(pri[i] * q * q);
}
// Z += zp*DT + SQDT*zdf*noise[t,j];  optional zs16 = Z / rowmax(Z). Real rows only.
__global__ void k_sde_update(float* __restrict__ Z, const float* __restrict__ zp,
                             const float* __restrict__ zdf,
                             const float* __restrict__ noise,
                             int t, int j, _Float16* __restrict__ zs) {
  __shared__ float sh[DST];
  int mr = blockIdx.x;               // 0 .. B*NN-1 (real rows)
  int b = mr / NN, n = mr % NN, d = threadIdx.x;
  long long i  = ((long long)b * NP + n) * DST + d;
  long long ni = ((((long long)t * 2 + j) * BB + b) * NN + n) * DST + d;
  float z = Z[i] + zp[i] * DT_ + SQDT * zdf[i] * noise[ni];
  Z[i] = z;
  if (zs) {
    sh[d] = z;
    __syncthreads();
    for (int s = 32; s > 0; s >>= 1) {
      if (d < s) sh[d] = fmaxf(sh[d], sh[d + s]);
      __syncthreads();
    }
    zs[i] = (_Float16)(z / sh[0]);
  }
}
// Wcomb = outW + [out_hW; 0] (f32), bcomb = outb + out_hb
__global__ void k_wcomb(float* __restrict__ wcb, float* __restrict__ bcb,
                        const float* __restrict__ outW, const float* __restrict__ outb,
                        const float* __restrict__ ohW, const float* __restrict__ ohb) {
  int i = blockIdx.x * blockDim.x + threadIdx.x;
  if (i < 192 * 64) {
    int k = i / 64, c = i % 64;
    wcb[i] = outW[i] + (k < 128 ? ohW[k * 64 + c] : 0.0f);
  }
  if (i < 64) bcb[i] = outb[i] + ohb[i];
}
// FIN16 = [h_pre_g(128) | zs(64)]
__global__ void k_packfin(_Float16* __restrict__ fin, const _Float16* __restrict__ hpg,
                          const _Float16* __restrict__ zs) {
  int i = blockIdx.x * blockDim.x + threadIdx.x;
  if (i >= MP * 192) return;
  int m = i / 192, d = i % 192;
  fin[i] = (d < 128) ? hpg[(long long)m * 128 + d]
                     : zs[(long long)m * 64 + (d - 128)];
}

// ---------------------------------------------------------------------------
extern "C" void kernel_launch(void* const* d_in, const int* in_sizes, int n_in,
                              void* d_out, int out_size, void* d_ws, size_t ws_size,
                              hipStream_t stream)
{
  (void)in_sizes; (void)n_in; (void)out_size; (void)ws_size;
  auto P = [&](int i) { return (const float*)d_in[i]; };

  const float* values = P(0);
  const float* masks  = P(1);
  const float* Adj    = P(3);
  const float* noise  = P(4);
  // params: jax pytree (alphabetical dict) order, leaves 5..67
  const float* pre_Whh = P(13); const float* pre_Wih = P(14);
  const float* pre_bhh = P(15); const float* pre_bih = P(16);
  const float* gg_cW = P(17); const float* gg_cB = P(18);
  const float* gg_gW = P(19); const float* gg_gB = P(20);
  const float* gv_cW = P(21); const float* gv_cB = P(22);
  const float* gv_gW = P(23); const float* gv_gB = P(24);
  const float* h0  = P(25);
  const float* difW[3] = { P(26), P(28), P(30) };
  const float* difB[3] = { P(27), P(29), P(31) };
  const float* drfW[3] = { P(32), P(34), P(36) };
  const float* drfB[3] = { P(33), P(35), P(37) };
  const float* priW[3] = { P(38), P(40), P(42) };
  const float* priB[3] = { P(39), P(41), P(43) };
  const float* odegW[3] = { P(44), P(46), P(48) };
  const float* odegB[3] = { P(45), P(47), P(49) };
  const float* odevW[3] = { P(50), P(52), P(54) };
  const float* odevB[3] = { P(51), P(53), P(55) };
  const float* outW = P(56); const float* outb = P(57);
  const float* ohW  = P(58); const float* ohb  = P(59);
  const float* z0s  = P(60); const float* z0v  = P(61);
  const float* zoW[3] = { P(62), P(64), P(66) };
  const float* zoB[3] = { P(63), P(65), P(67) };

  // ---------------- workspace (bump allocator, bytes) ----------------
  char* base = (char*)d_ws;
  size_t off = 0;
  auto allocF = [&](size_t n) { float* p = (float*)(base + off);
                                off += ((n * 4 + 255) & ~(size_t)255); return p; };
  auto allocH = [&](size_t n) { _Float16* p = (_Float16*)(base + off);
                                off += ((n * 2 + 255) & ~(size_t)255); return p; };

  // f16 activation buffers
  _Float16* HPG16 = allocH((size_t)TT * MP * DRNN);
  _Float16* ZE16  = allocH((size_t)TT * MP * DRNN);
  _Float16* ZS16  = allocH((size_t)TT * MP * DST);
  _Float16* XC_A  = allocH((size_t)MP * 576);
  _Float16* XC_B  = allocH((size_t)MP * 576);
  _Float16* XTA   = allocH((size_t)BB * 192 * NP);
  _Float16* XTB   = allocH((size_t)BB * 192 * NP);
  _Float16* ZV1_16 = allocH((size_t)MP * DRNN);
  _Float16* G1_16 = allocH((size_t)MP * DRNN);
  _Float16* G2_16 = allocH((size_t)MP * DRNN);
  _Float16* HST16 = allocH((size_t)MP * DRNN);
  _Float16* SH1_16 = allocH((size_t)MP * 128);
  _Float16* SH2_16 = allocH((size_t)MP * 128);
  _Float16* SIN16 = allocH((size_t)MP * DST);
  _Float16* FIN16 = allocH((size_t)MP * 192);
  // f32 buffers
  float* HV  = allocF((size_t)MP * DRNN); float* ZV  = allocF((size_t)MP * DRNN);
  float* HV1 = allocF((size_t)MP * DRNN); float* ZV1 = allocF((size_t)MP * DRNN);
  float* HVP = allocF((size_t)MP * DRNN); float* ZVP = allocF((size_t)MP * DRNN);
  float* RU  = allocF((size_t)MP * 256);  float* CC  = allocF((size_t)MP * DRNN);
  float* GT  = allocF((size_t)MP * DRNN); float* OM  = allocF((size_t)MP);
  float* GI  = allocF((size_t)MP * 384);  float* GH  = allocF((size_t)MP * 384);
  float* HST = allocF((size_t)MP * DRNN);
  float* PRI = allocF((size_t)MP * DST);  float* ZDF = allocF((size_t)MP * DST);
  float* ZP  = allocF((size_t)MP * DST);  float* ZST = allocF((size_t)MP * DST);
  float* WCB = allocF(192 * 64);          float* BCB = allocF(64);
  // f16 weights (pre-converted, col-major where needed)
  _Float16* AdjH = allocH((size_t)NP * NP);
  _Float16* odevT[3], *odegT[3], *zoT[3], *priT[3], *difT[3], *drfT[3];
  for (int i = 0; i < 3; ++i) { odevT[i] = allocH(384 * 128); odegT[i] = allocH(384 * 128); }
  for (int i = 0; i < 3; ++i) zoT[i] = allocH(128 * 128);
  priT[0] = allocH(128 * 128); priT[1] = allocH(128 * 128); priT[2] = allocH(128 * 64);
  difT[0] = allocH(128 * 128); difT[1] = allocH(128 * 128); difT[2] = allocH(128 * 64);
  drfT[0] = allocH(64 * 128);  drfT[1] = allocH(128 * 128); drfT[2] = allocH(128 * 64);
  _Float16* gvgT = allocH(576 * 256); _Float16* gvcT = allocH(576 * 128);
  _Float16* gggT = allocH(576 * 256); _Float16* ggcT = allocH(576 * 128);
  _Float16* WihC = allocH(384 * 128); _Float16* WhhC = allocH(384 * 128);
  _Float16* WCBT = allocH(192 * 64);

  // ---------------- launch helpers ----------------
  auto gemm = [&](const _Float16* A, int lda, long long sA,
                  const _Float16* Bt, int ldb, long long sB,
                  const float* bias,
                  const float* R, int ldr, long long sR, float rs,
                  float* D, int ldd, long long sD, int Mstore,
                  _Float16* R16, int ld16, long long s16,
                  _Float16* C16, int ldc, int cPer, long long sCz, long long sCm,
                  int M, int Nn, int Kk, int act, float osc, int batch) {
    dim3 g(Nn / 64, M / 16, batch);
    k_gemm<<<g, 32, 0, stream>>>(A, lda, sA, Bt, ldb, sB, bias, R, ldr, sR, rs,
                                 D, ldd, sD, Mstore, R16, ld16, s16,
                                 C16, ldc, cPer, sCz, sCm, Kk, act, osc);
  };
  auto EL = [&](int n) { return dim3((n + 255) / 256); };
  auto w16t = [&](_Float16* dst, const float* src, int K, int N) {
    k_w16t<<<EL(K * N), 256, 0, stream>>>(dst, src, K, N);
  };
  // diffusion hops: XC[:,din:2din] = A@slot0 (via XTs), XC[:,2din:3din] = A@slot1
  auto hops = [&](int din, _Float16* XC, int ld, _Float16* XTs, _Float16* XTd) {
    gemm(AdjH, NP, 0, XTs, NP, (long long)din * NP, nullptr,
         nullptr, 0, 0, 0.f, nullptr, 0, 0, 0,
         XC + din, ld, (long long)NP * ld,
         XTd, NP, NP, (long long)din * NP, 0,
         NP, din, NP, 0, 1.0f, BB);
    gemm(AdjH, NP, 0, XTd, NP, (long long)din * NP, nullptr,
         nullptr, 0, 0, 0.f, nullptr, 0, 0, 0,
         XC + 2 * din, ld, (long long)NP * ld,
         nullptr, 0, 1, 0, 0,
         NP, din, NP, 0, 1.0f, BB);
  };
  // dcode: out = res + DT*tanh(gconv(tanh(gconv(tanh(gconv(in))))))
  auto dcode = [&](const float* in, _Float16* const* Wt, const float* const* bl,
                   const float* res, float* out, _Float16* out16) {
    k_pack_h<<<EL(MP * DRNN), 256, 0, stream>>>(XC_A, 384, XTA, in);
    hops(128, XC_A, 384, XTA, XTB);
    gemm(XC_A, 384, 0, Wt[0], 384, 0, bl[0],
         nullptr, 0, 0, 0.f, nullptr, 0, 0, 0,
         XC_B, 384, 0,
         XTA, NP, NP, 0, (long long)128 * NP,
         MP, 128, 384, 1, 1.0f, 1);
    hops(128, XC_B, 384, XTA, XTB);
    gemm(XC_B, 384, 0, Wt[1], 384, 0, bl[1],
         nullptr, 0, 0, 0.f, nullptr, 0, 0, 0,
         XC_A, 384, 0,
         XTA, NP, NP, 0, (long long)128 * NP,
         MP, 128, 384, 1, 1.0f, 1);
    hops(128, XC_A, 384, XTA, XTB);
    gemm(XC_A, 384, 0, Wt[2], 384, 0, bl[2],
         res, 128, 0, 1.0f,
         out, 128, 0, MP,
         out16, 128, 0,
         nullptr, 0, 1, 0, 0,
         MP, 128, 384, 1, DT_, 1);
  };
  // diffusion-conv GRU cell blended with observation mask
  auto dcgru = [&](int t, const float* h, _Float16* gWT, const float* gB,
                   _Float16* cWT, const float* cB, float* outp) {
    k_pack_xh<<<EL(MP * 192), 256, 0, stream>>>(XC_A, 576, XTA, values, masks, h, t);
    hops(192, XC_A, 576, XTA, XTB);
    gemm(XC_A, 576, 0, gWT, 576, 0, gB, nullptr, 0, 0, 0.f,
         RU, 256, 0, MP, nullptr, 0, 0, nullptr, 0, 1, 0, 0,
         MP, 256, 576, 2, 1.0f, 1);
    k_pack_xrh<<<EL(MP * 192), 256, 0, stream>>>(XC_A, 576, XTA, values, masks, RU, h, t);
    hops(192, XC_A, 576, XTA, XTB);
    gemm(XC_A, 576, 0, cWT, 576, 0, cB, nullptr, 0, 0, 0.f,
         CC, 128, 0, MP, nullptr, 0, 0, nullptr, 0, 1, 0, 0,
         MP, 128, 576, 1, 1.0f, 1);
    k_gru_blend<<<EL(MP * DRNN), 256, 0, stream>>>(outp, h, RU, CC, OM);
  };
  // 3-layer tanh MLP (DensNet), output dim 64 f32
  auto mlp3 = [&](const _Float16* x16, int dinX, _Float16* const* Wt,
                  const float* const* bm, float* outF) {
    gemm(x16, dinX, 0, Wt[0], dinX, 0, bm[0], nullptr, 0, 0, 0.f,
         nullptr, 0, 0, 0, SH1_16, 128, 0, nullptr, 0, 1, 0, 0,
         MP, 128, dinX, 1, 1.0f, 1);
    gemm(SH1_16, 128, 0, Wt[1], 128, 0, bm[1], nullptr, 0, 0, 0.f,
         nullptr, 0, 0, 0, SH2_16, 128, 0, nullptr, 0, 1, 0, 0,
         MP, 128, 128, 1, 1.0f, 1);
    gemm(SH2_16, 128, 0, Wt[2], 128, 0, bm[2], nullptr, 0, 0, 0.f,
         outF, 64, 0, MP, nullptr, 0, 0, nullptr, 0, 1, 0, 0,
         MP, 64, 128, 1, 1.0f, 1);
  };

  // ================= phase 0: one-time weight conversion =================
  k_adj16<<<EL(NP * NP), 256, 0, stream>>>(AdjH, Adj);
  for (int i = 0; i < 3; ++i) { w16t(odevT[i], odevW[i], 384, 128);
                                w16t(odegT[i], odegW[i], 384, 128); }
  for (int i = 0; i < 3; ++i) w16t(zoT[i], zoW[i], 128, 128);
  w16t(priT[0], priW[0], 128, 128); w16t(priT[1], priW[1], 128, 128);
  w16t(priT[2], priW[2], 128, 64);
  w16t(difT[0], difW[0], 128, 128); w16t(difT[1], difW[1], 128, 128);
  w16t(difT[2], difW[2], 128, 64);
  w16t(drfT[0], drfW[0], 64, 128);  w16t(drfT[1], drfW[1], 128, 128);
  w16t(drfT[2], drfW[2], 128, 64);
  w16t(gvgT, gv_gW, 576, 256); w16t(gvcT, gv_cW, 576, 128);
  w16t(gggT, gg_gW, 576, 256); w16t(ggcT, gg_cW, 576, 128);
  k_w16c<<<EL(384 * 128), 256, 0, stream>>>(WihC, pre_Wih, 384 * 128);
  k_w16c<<<EL(384 * 128), 256, 0, stream>>>(WhhC, pre_Whh, 384 * 128);
  k_wcomb<<<EL(192 * 64), 256, 0, stream>>>(WCB, BCB, outW, outb, ohW, ohb);
  w16t(WCBT, WCB, 192, 64);

  // ================= phase 1: ODE/GRU scan over T =================
  k_broadcast<<<EL(MP * DRNN), 256, 0, stream>>>(HV, h0, MP, DRNN);
  k_broadcast<<<EL(MP * DRNN), 256, 0, stream>>>(ZV, z0v, MP, DRNN);

  for (int t = 0; t < TT; ++t) {
    dcode(HV, odevT, odevB, HV, HV1, nullptr);         // hv1
    dcode(ZV, odegT, odegB, ZV, ZV1, ZV1_16);          // zv1 (+f16 for gate)
    k_om<<<EL(MP), 256, 0, stream>>>(OM, masks, t);
    // gate(zv1); gated h_pre stored directly as f16
    gemm(ZV1_16, 128, 0, zoT[0], 128, 0, zoB[0], nullptr, 0, 0, 0.f,
         nullptr, 0, 0, 0, G1_16, 128, 0, nullptr, 0, 1, 0, 0,
         MP, 128, 128, 1, 1.0f, 1);
    gemm(G1_16, 128, 0, zoT[1], 128, 0, zoB[1], nullptr, 0, 0, 0.f,
         nullptr, 0, 0, 0, G2_16, 128, 0, nullptr, 0, 1, 0, 0,
         MP, 128, 128, 1, 1.0f, 1);
    gemm(G2_16, 128, 0, zoT[2], 128, 0, zoB[2], nullptr, 0, 0, 0.f,
         GT, 128, 0, MP, nullptr, 0, 0, nullptr, 0, 1, 0, 0,
         MP, 128, 128, 2, 1.0f, 1);
    k_mul16<<<EL(MP * DRNN), 256, 0, stream>>>(
        HPG16 + (long long)t * MP * DRNN, HV1, GT, MP * DRNN);
    // observation updates
    dcgru(t, HV1, gvgT, gv_gB, gvcT, gv_cB, HVP);
    dcgru(t, ZV1, gggT, gg_gB, ggcT, gg_cB, ZVP);
    if (t < TT - 1) {                                   // carry for next frame
      dcode(HVP, odevT, odevB, HVP, HV, nullptr);
      dcode(ZVP, odegT, odegB, ZVP, ZV, nullptr);
    }
  }

  // ================= phase 2: enc_pre GRU over T =================
  k_fill32<<<EL(MP * DRNN), 256, 0, stream>>>(HST, 0.0f, MP * DRNN);
  k_fill16<<<EL(MP * DRNN), 256, 0, stream>>>(HST16, MP * DRNN);
  for (int t = 0; t < TT; ++t) {
    gemm(HPG16 + (long long)t * MP * DRNN, 128, 0, WihC, 128, 0, pre_bih,
         nullptr, 0, 0, 0.f, GI, 384, 0, MP, nullptr, 0, 0, nullptr, 0, 1, 0, 0,
         MP, 384, 128, 0, 1.0f, 1);
    gemm(HST16, 128, 0, WhhC, 128, 0, pre_bhh,
         nullptr, 0, 0, 0.f, GH, 384, 0, MP, nullptr, 0, 0, nullptr, 0, 1, 0, 0,
         MP, 384, 128, 0, 1.0f, 1);
    k_gru_cell<<<EL(MP * DRNN), 256, 0, stream>>>(
        HST, HST16, ZE16 + (long long)t * MP * DRNN, GI, GH);
  }

  // ================= phase 3: SDE scan (samples identical -> run once) ======
  k_broadcast<<<EL(MP * DST), 256, 0, stream>>>(ZST, z0s, MP, DST);
  for (int t = 0; t < TT; ++t) {
    const _Float16* ze = ZE16 + (long long)t * MP * DRNN;
    mlp3(ze, 128, priT, priB, PRI);
    mlp3(ze, 128, difT, difB, ZDF);
    k_sde_in<<<EL(MP * DST), 256, 0, stream>>>(SIN16, PRI, ZDF, ZST, MP * DST);
    mlp3(SIN16, 64, drfT, drfB, ZP);
    k_sde_update<<<BB * NN, DST, 0, stream>>>(ZST, ZP, ZDF, noise, t, 0,
                                              ZS16 + (long long)t * MP * DST);
    k_sde_in<<<EL(MP * DST), 256, 0, stream>>>(SIN16, PRI, ZDF, ZST, MP * DST);
    mlp3(SIN16, 64, drfT, drfB, ZP);
    k_sde_update<<<BB * NN, DST, 0, stream>>>(ZST, ZP, ZDF, noise, t, 1, nullptr);
  }

  // ================= phase 4: fused output projection =================
  // out[b,t-1,n,:] = [h_pre_g | Zs] @ (Wf + [Wo;0]) + (bf + bo)
  float* OUT = (float*)d_out;
  for (int t = 1; t < TT; ++t) {
    k_packfin<<<EL(MP * 192), 256, 0, stream>>>(
        FIN16, HPG16 + (long long)t * MP * DRNN, ZS16 + (long long)t * MP * DST);
    gemm(FIN16, 192, (long long)NP * 192, WCBT, 192, 0, BCB,
         nullptr, 0, 0, 0.f,
         OUT + (long long)(t - 1) * NN * DIN, 64, (long long)(TT - 1) * NN * DIN, NN,
         nullptr, 0, 0, nullptr, 0, 1, 0, 0,
         NP, 64, 192, 0, 1.0f, BB);
  }
}
